// TransactionGraphModel_11871289606411
// MI455X (gfx1250) — matrix operand order repaired and checked by reading, not compiled, not run on hardware
//
#include <hip/hip_runtime.h>
#include <hip/hip_bf16.h>

typedef __attribute__((ext_vector_type(16))) __bf16 v16bf;
typedef __attribute__((ext_vector_type(8)))  __bf16 v8bf;
typedef __attribute__((ext_vector_type(4)))  __bf16 v4bf;
typedef __attribute__((ext_vector_type(8)))  float  v8f;

#define IN_DIM  128
#define HID_DIM 128
#define OUT_DIM 64
#define BPITCH  136   // bf16 pitch: 272B rows -> successive rows 4 banks apart

// ---------------------------------------------------------------- helpers

__device__ __forceinline__ v8f wmma_bf16(v16bf a, v16bf b, v8f c) {
  // (neg_a, A, neg_b, B, c_mod, C, reuse_a, reuse_b)
  return __builtin_amdgcn_wmma_f32_16x16x32_bf16(false, a, false, b,
                                                 (short)0, c, false, false);
}

__device__ __forceinline__ void atomAddF(float* p, float v) {
  unsafeAtomicAdd(p, v);   // hardware global_atomic_add_f32 (no-return)
}

// A fragment (16x32 bf16, MxK): lane&15 = M row.
// lanes 0-15:  elems 0-15 -> K=kb+0..15
// lanes 16-31: elems 0-15 -> K=kb+16..31
__device__ __forceinline__ v16bf load_A(const __bf16* __restrict__ srow,
                                        int kb, int hi) {
  v8bf lo = *(const v8bf*)(srow + kb + hi * 16);       // 16B ds_load_b128
  v8bf up = *(const v8bf*)(srow + kb + 8 + hi * 16);   // 16B ds_load_b128
  return __builtin_shufflevector(lo, up, 0, 1, 2, 3, 4, 5, 6, 7,
                                         8, 9, 10, 11, 12, 13, 14, 15);
}

// B fragment (32x16 bf16, KxN): lane&15 = N col; lanes 0-15 hold K=kb+0..15,
// lanes 16-31 hold K=kb+16..31.  B[k][n] = W[n][k]  (W row-major [out, in])
__device__ __forceinline__ v16bf load_B(const __bf16* __restrict__ wrow,
                                        int kb, int hi) {
  return *(const v16bf*)(wrow + kb + hi * 16);         // 32B vector load
}

__device__ __forceinline__ v4bf pack4(float a, float b, float c, float d) {
  v4bf p;
  p[0] = (__bf16)a; p[1] = (__bf16)b; p[2] = (__bf16)c; p[3] = (__bf16)d;
  return p;
}

// ---------------------------------------------------------------- kernels

__global__ void k_zero(float* __restrict__ p, long n) {
  long i = (long)blockIdx.x * blockDim.x + threadIdx.x;
  long stride = (long)gridDim.x * blockDim.x;
  for (; i < n; i += stride) p[i] = 0.0f;
}

__global__ void k_cvtw(const float* __restrict__ w, __bf16* __restrict__ o,
                       int n) {
  int i = blockIdx.x * blockDim.x + threadIdx.x;
  if (i < n) o[i] = (__bf16)w[i];
}

__global__ void k_gcnt(const int* __restrict__ batch,
                       float* __restrict__ gcnt, int n) {
  int i = blockIdx.x * blockDim.x + threadIdx.x;
  if (i < n) atomAddF(&gcnt[batch[i]], 1.0f);
}

// one wave per edge; lane moves 4 consecutive feats (32*4 = 128)
__global__ void __launch_bounds__(256) k_scatter(
    const float* __restrict__ feat, const int* __restrict__ src,
    const int* __restrict__ dst, float* __restrict__ agg,
    float* __restrict__ cnt, int nEdges) {
  int e = (blockIdx.x * blockDim.x + threadIdx.x) >> 5;
  int lane = threadIdx.x & 31;
  if (e >= nEdges) return;
  int s = src[e];
  int d = dst[e];
  float4 v = ((const float4*)(feat + (long)s * IN_DIM))[lane];
  float* arow = agg + (long)d * IN_DIM + lane * 4;
  atomAddF(arow + 0, v.x);
  atomAddF(arow + 1, v.y);
  atomAddF(arow + 2, v.z);
  atomAddF(arow + 3, v.w);
  if (cnt != nullptr && lane == 0) atomAddF(&cnt[d], 1.0f);
}

// layer 1: h1 = relu( mean_agg @ Wl^T + x @ Wr^T + b ), 16-node tile x 128
__global__ void __launch_bounds__(256) k_sage1(
    const float* __restrict__ x, const float* __restrict__ agg,
    const float* __restrict__ cnt, const __bf16* __restrict__ Wl,
    const __bf16* __restrict__ Wr, const float* __restrict__ bias,
    float* __restrict__ h1, int nNodes) {
  __shared__ __align__(16) __bf16 sA[16 * BPITCH];   // mean aggregate (bf16)
  __shared__ __align__(16) __bf16 sX[16 * BPITCH];   // self features  (bf16)
  const long base = (long)blockIdx.x * 16;
  const int tid = threadIdx.x;

  for (int i = tid; i < 16 * 32; i += 256) {   // float4 granules
    int row = i >> 5;
    int c4  = i & 31;
    long node = base + row;
    if (node >= nNodes) node = nNodes - 1;
    float4 a = ((const float4*)(agg + node * IN_DIM))[c4];
    float inv = 1.0f / fmaxf(cnt[node], 1.0f);
    *(v4bf*)(sA + row * BPITCH + c4 * 4) =
        pack4(a.x * inv, a.y * inv, a.z * inv, a.w * inv);
    float4 xv = ((const float4*)(x + node * IN_DIM))[c4];
    *(v4bf*)(sX + row * BPITCH + c4 * 4) = pack4(xv.x, xv.y, xv.z, xv.w);
  }
  __syncthreads();

  const int ct   = tid >> 5;        // column tile 0..7
  const int lane = tid & 31;
  const int ln15 = lane & 15;
  const int hi   = lane >> 4;
  const int ncol = ct * 16 + ln15;  // output channel
  const __bf16* aRow  = sA + ln15 * BPITCH;
  const __bf16* xRow  = sX + ln15 * BPITCH;
  const __bf16* wlRow = Wl + (long)ncol * IN_DIM;
  const __bf16* wrRow = Wr + (long)ncol * IN_DIM;

  v8f c = {};
#pragma unroll
  for (int kb = 0; kb < IN_DIM; kb += 32) {
    c = wmma_bf16(load_A(aRow, kb, hi), load_B(wlRow, kb, hi), c);
    c = wmma_bf16(load_A(xRow, kb, hi), load_B(wrRow, kb, hi), c);
  }

  const float bv = bias[ncol];
#pragma unroll
  for (int r = 0; r < 8; ++r) {
    long node = base + r + hi * 8;
    if (node < nNodes)
      h1[node * HID_DIM + ncol] = fmaxf(c[r] + bv, 0.0f);
  }
}

// layer 2 + fused mean pool: out[g] += (row + b2) / gcnt[g]
__global__ void __launch_bounds__(128) k_sage2(
    const float* __restrict__ h1, const float* __restrict__ agg,
    const float* __restrict__ cnt, const __bf16* __restrict__ Wl,
    const __bf16* __restrict__ Wr, const float* __restrict__ bias,
    const int* __restrict__ batch, const float* __restrict__ gcnt,
    float* __restrict__ out, int nNodes) {
  __shared__ __align__(16) __bf16 sA[16 * BPITCH];
  __shared__ __align__(16) __bf16 sX[16 * BPITCH];
  const long base = (long)blockIdx.x * 16;
  const int tid = threadIdx.x;

  for (int i = tid; i < 16 * 32; i += 128) {
    int row = i >> 5;
    int c4  = i & 31;
    long node = base + row;
    if (node >= nNodes) node = nNodes - 1;
    float4 a = ((const float4*)(agg + node * HID_DIM))[c4];
    float inv = 1.0f / fmaxf(cnt[node], 1.0f);
    *(v4bf*)(sA + row * BPITCH + c4 * 4) =
        pack4(a.x * inv, a.y * inv, a.z * inv, a.w * inv);
    float4 xv = ((const float4*)(h1 + node * HID_DIM))[c4];
    *(v4bf*)(sX + row * BPITCH + c4 * 4) = pack4(xv.x, xv.y, xv.z, xv.w);
  }
  __syncthreads();

  const int ct   = tid >> 5;        // column tile 0..3 (OUT_DIM = 64)
  const int lane = tid & 31;
  const int ln15 = lane & 15;
  const int hi   = lane >> 4;
  const int ncol = ct * 16 + ln15;
  const __bf16* aRow  = sA + ln15 * BPITCH;
  const __bf16* xRow  = sX + ln15 * BPITCH;
  const __bf16* wlRow = Wl + (long)ncol * HID_DIM;
  const __bf16* wrRow = Wr + (long)ncol * HID_DIM;

  v8f c = {};
#pragma unroll
  for (int kb = 0; kb < HID_DIM; kb += 32) {
    c = wmma_bf16(load_A(aRow, kb, hi), load_B(wlRow, kb, hi), c);
    c = wmma_bf16(load_A(xRow, kb, hi), load_B(wrRow, kb, hi), c);
  }

  const float bv = bias[ncol];
#pragma unroll
  for (int r = 0; r < 8; ++r) {
    long node = base + r + hi * 8;
    if (node < nNodes) {
      int g = batch[node];
      float s = 1.0f / fmaxf(gcnt[g], 1.0f);
      atomAddF(&out[(long)g * OUT_DIM + ncol], (c[r] + bv) * s);
    }
  }
}

// ---------------------------------------------------------------- launch

extern "C" void kernel_launch(void* const* d_in, const int* in_sizes, int n_in,
                              void* d_out, int out_size, void* d_ws,
                              size_t ws_size, hipStream_t stream) {
  const float* x     = (const float*)d_in[0];
  const int*   edge  = (const int*)d_in[1];
  const int*   batch = (const int*)d_in[2];
  const float* Wl1   = (const float*)d_in[3];
  const float* Wr1   = (const float*)d_in[4];
  const float* b1    = (const float*)d_in[5];
  const float* Wl2   = (const float*)d_in[6];
  const float* Wr2   = (const float*)d_in[7];
  const float* b2    = (const float*)d_in[8];

  const int nNodes  = in_sizes[0] / IN_DIM;
  const int nEdges  = in_sizes[1] / 2;
  const int nGraphs = out_size / OUT_DIM;
  const int* src = edge;
  const int* dst = edge + nEdges;

  // ws layout: agg | cnt | gcnt | h1 | bf16 weights   (agg reused per layer)
  float* agg  = (float*)d_ws;
  float* cnt  = agg + (size_t)nNodes * IN_DIM;
  float* gcnt = cnt + nNodes;
  float* h1   = gcnt + nGraphs;
  __bf16* Wl1b = (__bf16*)(h1 + (size_t)nNodes * HID_DIM);
  __bf16* Wr1b = Wl1b + (size_t)HID_DIM * IN_DIM;
  __bf16* Wl2b = Wr1b + (size_t)HID_DIM * IN_DIM;
  __bf16* Wr2b = Wl2b + (size_t)OUT_DIM * HID_DIM;

  const long zeroN = (long)nNodes * IN_DIM + nNodes + nGraphs;
  k_zero<<<1024, 256, 0, stream>>>(agg, zeroN);
  k_zero<<<64, 256, 0, stream>>>((float*)d_out, (long)out_size);
  k_cvtw<<<(HID_DIM * IN_DIM + 255) / 256, 256, 0, stream>>>(Wl1, Wl1b,
                                                             HID_DIM * IN_DIM);
  k_cvtw<<<(HID_DIM * IN_DIM + 255) / 256, 256, 0, stream>>>(Wr1, Wr1b,
                                                             HID_DIM * IN_DIM);
  k_cvtw<<<(OUT_DIM * HID_DIM + 255) / 256, 256, 0, stream>>>(Wl2, Wl2b,
                                                              OUT_DIM * HID_DIM);
  k_cvtw<<<(OUT_DIM * HID_DIM + 255) / 256, 256, 0, stream>>>(Wr2, Wr2b,
                                                              OUT_DIM * HID_DIM);
  k_gcnt<<<(nNodes + 255) / 256, 256, 0, stream>>>(batch, gcnt, nNodes);

  const int sblocks = (int)(((long)nEdges * 32 + 255) / 256);
  const int ntiles  = (nNodes + 15) / 16;

  // layer 1
  k_scatter<<<sblocks, 256, 0, stream>>>(x, src, dst, agg, cnt, nEdges);
  k_sage1<<<ntiles, 256, 0, stream>>>(x, agg, cnt, Wl1b, Wr1b, b1, h1, nNodes);

  // layer 2 (degree counts unchanged; only re-zero agg)
  k_zero<<<1024, 256, 0, stream>>>(agg, (long)nNodes * IN_DIM);
  k_scatter<<<sblocks, 256, 0, stream>>>(h1, src, dst, agg, nullptr, nEdges);
  k_sage2<<<ntiles, 128, 0, stream>>>(h1, agg, cnt, Wl2b, Wr2b, b2, batch,
                                      gcnt, (float*)d_out, nNodes);
}
